// AttentionHead_32633161515671
// MI455X (gfx1250) — matrix operand order repaired and checked by reading, not compiled, not run on hardware
//
#include <hip/hip_runtime.h>
#include <hip/hip_bf16.h>

// ---------------------------------------------------------------------------
// Fused attention head for gfx1250 (MI455X), wave32 WMMA bf16 path.
//   Q = (x@Wq + bq) * 1/sqrt(128), K = x@Wk + bk, V = x@Wv + bv
//   out = softmax(Q@K^T) @ V
// Precision: bf16x2 split (hi+lo) for x,W,Q,K -> 3-term WMMA chains give
// near-fp32 scores; single-bf16 for V / attn@V.
// Data movement: double-buffered K/V tiles staged to LDS via
// global_load_async_to_lds_b128 (ASYNCcnt, partial waits for overlap) and
// TDM tensor_load_to_lds (TENSORcnt, hardware LDS row padding).
// Parallelism: 4-way KV split (flash-decoding) + combine pass.
// ---------------------------------------------------------------------------

constexpr int SEQ = 8192;
constexpr int DM  = 1024;
constexpr int DKK = 128;
constexpr int NSPLIT = 4;
constexpr int SEGLEN = SEQ / NSPLIT;             // 2048 keys per segment

typedef __attribute__((ext_vector_type(16))) __bf16 v16bf;
typedef __attribute__((ext_vector_type(8)))  __bf16 v8bf;
typedef __attribute__((ext_vector_type(8)))  float  v8f;
typedef __attribute__((ext_vector_type(4)))  unsigned int uint32x4;
typedef __attribute__((ext_vector_type(4)))  int int32x4;
typedef __attribute__((ext_vector_type(8)))  int int32x8;

union V16 { v16bf v; v8bf h[2]; __bf16 e[16]; };

#if __has_builtin(__builtin_amdgcn_tensor_load_to_lds)
#define HAVE_TDM 1
#else
#define HAVE_TDM 0
#endif

// workspace layout (bf16 elements), fp32 partials appended after
constexpr size_t OQH = 0;
constexpr size_t NSD = (size_t)SEQ * DKK;        // 1,048,576
constexpr size_t OQL = 1 * NSD;
constexpr size_t OKH = 2 * NSD;
constexpr size_t OKL = 3 * NSD;
constexpr size_t OVT = 4 * NSD;                  // V transposed [DKK][SEQ]
constexpr size_t OW  = 5 * NSD;                  // transposed/split weights
constexpr size_t WSZ = (size_t)DM * DKK;         // 131,072 per weight array
constexpr size_t BF_BYTES = (5 * NSD + 6 * WSZ) * 2;
constexpr size_t FP_OFF   = (BF_BYTES + 255) & ~(size_t)255;

__device__ __forceinline__ v8f wmma_bf16(v16bf a, v16bf b, v8f c) {
  return __builtin_amdgcn_wmma_f32_16x16x32_bf16(false, a, false, b,
                                                 (short)0, c, false, false);
}

__device__ __forceinline__ void splitf(float v, __bf16 &h, __bf16 &l) {
  h = (__bf16)v;
  l = (__bf16)(v - (float)h);
}

// 16B-per-lane async DMA: LDS[lds_off] = MEM[gbase + goff]  (ASYNCcnt)
__device__ __forceinline__ void async_copy16(unsigned int lds_off,
                                             const void* gbase,
                                             unsigned int goff) {
  asm volatile("global_load_async_to_lds_b128 %0, %1, %2"
               :: "v"(lds_off), "v"(goff), "s"(gbase)
               : "memory");
}

// --------------------------------------------------------------------------
// Kernel 0: transpose W [DM][DKK] -> Wt [DKK][DM], split into bf16 hi/lo.
// --------------------------------------------------------------------------
__global__ __launch_bounds__(256) void wsplit_kernel(const float* __restrict__ WQ,
                                                     const float* __restrict__ WK,
                                                     const float* __restrict__ WV,
                                                     __bf16* __restrict__ wsbf) {
  const int idx = blockIdx.x * 256 + threadIdx.x;      // over DM*DKK
  const int mat = blockIdx.y;
  const float* W = (mat == 0) ? WQ : (mat == 1) ? WK : WV;
  const int k = idx / DKK;
  const int n = idx % DKK;
  __bf16 h, l;
  splitf(W[idx], h, l);
  __bf16* base = wsbf + OW + (size_t)mat * 2 * WSZ;
  base[(size_t)n * DM + k]       = h;
  base[WSZ + (size_t)n * DM + k] = l;
}

// --------------------------------------------------------------------------
// Kernel 1: projections. block = 128 (4 waves); wave handles 2 n-tiles of a
// 16-row m-tile. 3-term bf16x2 WMMA, fp32 accumulate, bias, split-store.
// --------------------------------------------------------------------------
__global__ __launch_bounds__(128) void proj_kernel(const float* __restrict__ x,
                                                   const float* __restrict__ bQ,
                                                   const float* __restrict__ bK,
                                                   const float* __restrict__ bV,
                                                   __bf16* __restrict__ wsbf) {
  const int mat = blockIdx.y;
  const int m0  = blockIdx.x * 16;
  const int wv  = threadIdx.x >> 5;
  const int ln  = threadIdx.x & 31;
  const int lo  = ln & 15;
  const int hi  = ln >> 4;

  const __bf16* Bh = wsbf + OW + (size_t)mat * 2 * WSZ;
  const __bf16* Bl = Bh + WSZ;
  const float*  bias = (mat == 0) ? bQ : (mat == 1) ? bK : bV;

  const int n0a = (wv * 2) * 16;
  const int n0b = n0a + 16;

  const float* arow = x + (size_t)(m0 + lo) * DM;

  v8f c0 = {};
  v8f c1 = {};

  for (int k0 = 0; k0 < DM; k0 += 32) {
    const int ka = k0 + 16 * hi;
    float4 f0 = *(const float4*)(arow + ka);
    float4 f1 = *(const float4*)(arow + ka + 4);
    float4 f2 = *(const float4*)(arow + ka + 8);
    float4 f3 = *(const float4*)(arow + ka + 12);
    V16 ah, al;
#pragma unroll
    for (int i = 0; i < 4; ++i) {
      splitf(((const float*)&f0)[i], ah.e[i],      al.e[i]);
      splitf(((const float*)&f1)[i], ah.e[4 + i],  al.e[4 + i]);
      splitf(((const float*)&f2)[i], ah.e[8 + i],  al.e[8 + i]);
      splitf(((const float*)&f3)[i], ah.e[12 + i], al.e[12 + i]);
    }
    const size_t offA = (size_t)(n0a + lo) * DM + k0 + 16 * hi;
    const size_t offB = (size_t)(n0b + lo) * DM + k0 + 16 * hi;
    v16bf bh0 = *(const v16bf*)(Bh + offA);
    v16bf bl0 = *(const v16bf*)(Bl + offA);
    v16bf bh1 = *(const v16bf*)(Bh + offB);
    v16bf bl1 = *(const v16bf*)(Bl + offB);

    c0 = wmma_bf16(ah.v, bh0, c0);
    c0 = wmma_bf16(ah.v, bl0, c0);
    c0 = wmma_bf16(al.v, bh0, c0);

    c1 = wmma_bf16(ah.v, bh1, c1);
    c1 = wmma_bf16(ah.v, bl1, c1);
    c1 = wmma_bf16(al.v, bh1, c1);
  }

  const float scl = (mat == 0) ? 0.08838834764831845f : 1.0f;  // 1/sqrt(128)
  __bf16* Qh = wsbf + OQH;
  __bf16* Ql = wsbf + OQL;
  __bf16* Kh = wsbf + OKH;
  __bf16* Kl = wsbf + OKL;
  __bf16* Vt = wsbf + OVT;

#pragma unroll
  for (int t = 0; t < 2; ++t) {
    const int n0 = t ? n0b : n0a;
    const v8f  c  = t ? c1 : c0;
    const float bv = bias[n0 + lo];
#pragma unroll
    for (int r = 0; r < 8; ++r) {
      const int row = m0 + r + 8 * hi;
      const float val = (c[r] + bv) * scl;
      __bf16 h, l;
      splitf(val, h, l);
      if (mat == 0) {
        Qh[(size_t)row * DKK + n0 + lo] = h;
        Ql[(size_t)row * DKK + n0 + lo] = l;
      } else if (mat == 1) {
        Kh[(size_t)row * DKK + n0 + lo] = h;
        Kl[(size_t)row * DKK + n0 + lo] = l;
      } else {
        Vt[(size_t)(n0 + lo) * SEQ + row] = h;   // transposed store
      }
    }
  }
}

// --------------------------------------------------------------------------
// Kernel 2: flash attention over one KV segment per block (grid.y = split).
// block = 256 (8 waves), each wave one 16-row Q block. Double-buffered LDS
// staging: stage(i+1) issued before waiting for stage(i) -> DMA overlaps the
// 32 WMMAs of step i. Writes unnormalized partials (O_s, m_s, l_s).
// --------------------------------------------------------------------------
__global__ __launch_bounds__(256) void attn_kernel(const __bf16* __restrict__ wsbf,
                                                   float* __restrict__ pO,
                                                   float* __restrict__ pM,
                                                   float* __restrict__ pL) {
  __shared__ __attribute__((aligned(128))) __bf16 KhS[2][32][136]; // pad->272B rows
  __shared__ __attribute__((aligned(128))) __bf16 KlS[2][32][136];
  __shared__ __attribute__((aligned(128))) __bf16 VS[2][128][40];  // pad->80B rows
  __shared__ __attribute__((aligned(128))) __bf16 PS[8][16 * 40];

  const int tid = threadIdx.x;
  const int wv = tid >> 5;
  const int ln = tid & 31;
  const int lo = ln & 15;
  const int hi = ln >> 4;
  const int q0 = (blockIdx.x * 8 + wv) * 16;
  const int seg = blockIdx.y;
  const int kvbase = seg * SEGLEN;

  const __bf16* Qh  = wsbf + OQH;
  const __bf16* Ql  = wsbf + OQL;
  const __bf16* KhG = wsbf + OKH;
  const __bf16* KlG = wsbf + OKL;
  const __bf16* VtG = wsbf + OVT;

  const unsigned int khL = (unsigned int)(uintptr_t)(&KhS[0][0][0]);
  const unsigned int klL = (unsigned int)(uintptr_t)(&KlS[0][0][0]);
  const unsigned int vL  = (unsigned int)(uintptr_t)(&VS[0][0][0]);
  constexpr unsigned int KBUF = 32u * 136u * 2u;   // 8704 B per K buffer
  constexpr unsigned int VBUF = 128u * 40u * 2u;   // 10240 B per V buffer

  // ---- stage one 32-key step (buffer b): async K copies + TDM/async V ----
  auto stage = [&](int kv0, int b) {
#pragma unroll
    for (int i = 0; i < 2; ++i) {
      const int c = tid + 256 * i;
      const int row = c >> 4;
      const int part = c & 15;
      const unsigned int goff = (unsigned int)(kv0 + row) * 256u + part * 16u;
      const unsigned int loff = (unsigned int)b * KBUF +
                                (unsigned int)row * 272u + part * 16u;
      async_copy16(khL + loff, KhG, goff);
      async_copy16(klL + loff, KlG, goff);
    }
#if HAVE_TDM
    if (wv == 0) {
      // 2D tile: 32 elems (contig) x 128 rows (stride SEQ), 2B data;
      // TDM pads LDS by 4 DWORDs every 16 DWORDs -> 80B rows.
      const unsigned long long ga = (unsigned long long)(uintptr_t)(VtG + kv0);
      uint32x4 g0 = { 1u, vL + (unsigned int)b * VBUF,
                      (unsigned int)ga,
                      (unsigned int)((ga >> 32) & 0x01FFFFFFu) | 0x80000000u };
      int32x8 g1 = { (int)((1u << 16) | (1u << 20) | (3u << 22) | (3u << 25)),
                     (int)(32u << 16),         // tensor_dim0 = 32
                     (int)(128u << 16),        // tensor_dim1 = 128
                     (int)(32u << 16),         // tile_dim0 = 32
                     128,                      // tile_dim1 = 128
                     SEQ,                      // tensor_dim0_stride
                     0, 0 };
      int32x4 z4 = { 0, 0, 0, 0 };
#if __clang_major__ >= 23
      int32x8 z8 = { 0, 0, 0, 0, 0, 0, 0, 0 };
      __builtin_amdgcn_tensor_load_to_lds(g0, g1, z4, z4, z8, 0);
#else
      __builtin_amdgcn_tensor_load_to_lds(g0, g1, z4, z4, 0);
#endif
    }
#else
#pragma unroll
    for (int i = 0; i < 2; ++i) {
      const int c = tid + 256 * i;
      const int row = c >> 2;
      const int part = c & 3;
      const unsigned int goff =
          (unsigned int)row * (SEQ * 2u) + (unsigned int)kv0 * 2u + part * 16u;
      async_copy16(vL + (unsigned int)b * VBUF +
                   (unsigned int)row * 80u + part * 16u, VtG, goff);
    }
#endif
  };

  // ---- resident Q tile (16x128) hi/lo, A-layout, 4 K-chunks of 32
  v16bf qh[4], ql[4];
#pragma unroll
  for (int c = 0; c < 4; ++c) {
    const size_t off = (size_t)(q0 + lo) * DKK + 32 * c + 16 * hi;
    V16 th, tl;
    th.h[0] = *(const v8bf*)(Qh + off);
    th.h[1] = *(const v8bf*)(Qh + off + 8);
    tl.h[0] = *(const v8bf*)(Ql + off);
    tl.h[1] = *(const v8bf*)(Ql + off + 8);
    qh[c] = th.v;
    ql[c] = tl.v;
  }

  v8f acc[8];
  float m[8], lsum[8];
#pragma unroll
  for (int j = 0; j < 8; ++j) acc[j] = {};
#pragma unroll
  for (int r = 0; r < 8; ++r) { m[r] = -3.0e38f; lsum[r] = 0.0f; }

  __bf16* Pw = &PS[wv][0];
  constexpr float L2E = 1.4426950408889634f;
  constexpr int NSTEPS = SEGLEN / 32;              // 64

  stage(kvbase, 0);

  for (int i = 0; i < NSTEPS; ++i) {
    const int b = i & 1;
    if (i + 1 < NSTEPS) {
      stage(kvbase + (i + 1) * 32, b ^ 1);
      // wait only for stage(i): async loads complete in order
#if HAVE_TDM
      asm volatile("s_wait_asynccnt 4" ::: "memory");
      if (wv == 0) __builtin_amdgcn_s_wait_tensorcnt(1);
#else
      asm volatile("s_wait_asynccnt 6" ::: "memory");
#endif
    } else {
      asm volatile("s_wait_asynccnt 0" ::: "memory");
#if HAVE_TDM
      if (wv == 0) __builtin_amdgcn_s_wait_tensorcnt(0);
#endif
    }
    __syncthreads();

    // ================= scores: two 16x16 C tiles =========================
    v8f s0 = {};
    v8f s1 = {};
#pragma unroll
    for (int c = 0; c < 4; ++c) {
      const int co = 32 * c + 16 * hi;
      V16 bh0, bl0, bh1, bl1;
      bh0.h[0] = *(const v8bf*)(&KhS[b][lo][co]);
      bh0.h[1] = *(const v8bf*)(&KhS[b][lo][co + 8]);
      bl0.h[0] = *(const v8bf*)(&KlS[b][lo][co]);
      bl0.h[1] = *(const v8bf*)(&KlS[b][lo][co + 8]);
      bh1.h[0] = *(const v8bf*)(&KhS[b][16 + lo][co]);
      bh1.h[1] = *(const v8bf*)(&KhS[b][16 + lo][co + 8]);
      bl1.h[0] = *(const v8bf*)(&KlS[b][16 + lo][co]);
      bl1.h[1] = *(const v8bf*)(&KlS[b][16 + lo][co + 8]);

      s0 = wmma_bf16(qh[c], bh0.v, s0);
      s0 = wmma_bf16(qh[c], bl0.v, s0);
      s0 = wmma_bf16(ql[c], bh0.v, s0);

      s1 = wmma_bf16(qh[c], bh1.v, s1);
      s1 = wmma_bf16(qh[c], bl1.v, s1);
      s1 = wmma_bf16(ql[c], bh1.v, s1);
    }

    // ================= online softmax (per-lane rows r+8*hi) =============
    float sc[8];
#pragma unroll
    for (int r = 0; r < 8; ++r) {
      float v = fmaxf(s0[r], s1[r]);
      v = fmaxf(v, __shfl_xor(v, 1, 32));
      v = fmaxf(v, __shfl_xor(v, 2, 32));
      v = fmaxf(v, __shfl_xor(v, 4, 32));
      v = fmaxf(v, __shfl_xor(v, 8, 32));
      const float mn = fmaxf(m[r], v);
      sc[r] = __builtin_amdgcn_exp2f((m[r] - mn) * L2E);
      m[r] = mn;
      const float p0 = __builtin_amdgcn_exp2f((s0[r] - mn) * L2E);
      const float p1 = __builtin_amdgcn_exp2f((s1[r] - mn) * L2E);
      s0[r] = p0;
      s1[r] = p1;
      float ps = p0 + p1;
      ps += __shfl_xor(ps, 1, 32);
      ps += __shfl_xor(ps, 2, 32);
      ps += __shfl_xor(ps, 4, 32);
      ps += __shfl_xor(ps, 8, 32);
      lsum[r] = lsum[r] * sc[r] + ps;
    }

#pragma unroll
    for (int j = 0; j < 8; ++j)
#pragma unroll
      for (int r = 0; r < 8; ++r) acc[j][r] *= sc[r];

    // ================= P (C-layout) -> LDS (rows padded to 40) ===========
#pragma unroll
    for (int r = 0; r < 8; ++r) {
      const int row = r + 8 * hi;
      Pw[row * 40 + lo]      = (__bf16)s0[r];
      Pw[row * 40 + 16 + lo] = (__bf16)s1[r];
    }
    asm volatile("s_wait_dscnt 0x0" ::: "memory");

    V16 pa;
    {
      const __bf16* p = Pw + lo * 40 + 16 * hi;
      pa.h[0] = *(const v8bf*)p;
      pa.h[1] = *(const v8bf*)(p + 8);
    }

    // ================= O += P @ V ========================================
#pragma unroll
    for (int j = 0; j < 8; ++j) {
      V16 vb;
      vb.h[0] = *(const v8bf*)(&VS[b][16 * j + lo][16 * hi]);
      vb.h[1] = *(const v8bf*)(&VS[b][16 * j + lo][16 * hi + 8]);
      acc[j] = wmma_bf16(pa.v, vb.v, acc[j]);
    }

    __syncthreads();   // reads done before stage(i+2) overwrites buffer b
  }

  // ---- write unnormalized partials for this segment
#pragma unroll
  for (int j = 0; j < 8; ++j) {
#pragma unroll
    for (int r = 0; r < 8; ++r) {
      const size_t row = (size_t)seg * SEQ + q0 + r + 8 * hi;
      pO[row * DKK + 16 * j + lo] = acc[j][r];
    }
  }
  if (lo == 0) {
#pragma unroll
    for (int r = 0; r < 8; ++r) {
      const size_t row = (size_t)seg * SEQ + q0 + r + 8 * hi;
      pM[row] = m[r];
      pL[row] = lsum[r];
    }
  }
}

// --------------------------------------------------------------------------
// Kernel 3: combine the NSPLIT partials per output element.
// --------------------------------------------------------------------------
__global__ __launch_bounds__(256) void combine_kernel(const float* __restrict__ pO,
                                                      const float* __restrict__ pM,
                                                      const float* __restrict__ pL,
                                                      float* __restrict__ out) {
  const int idx = blockIdx.x * 256 + threadIdx.x;   // over SEQ*DKK
  const int row = idx / DKK;
  constexpr float L2E = 1.4426950408889634f;

  float mm = -3.0e38f;
#pragma unroll
  for (int s = 0; s < NSPLIT; ++s)
    mm = fmaxf(mm, pM[(size_t)s * SEQ + row]);

  float num = 0.0f, den = 0.0f;
#pragma unroll
  for (int s = 0; s < NSPLIT; ++s) {
    const size_t srow = (size_t)s * SEQ + row;
    const float w = __builtin_amdgcn_exp2f((pM[srow] - mm) * L2E);
    num += w * pO[srow * DKK + (idx % DKK)];
    den += w * pL[srow];
  }
  out[idx] = num / den;
}

// --------------------------------------------------------------------------
extern "C" void kernel_launch(void* const* d_in, const int* in_sizes, int n_in,
                              void* d_out, int out_size, void* d_ws, size_t ws_size,
                              hipStream_t stream) {
  const float* x  = (const float*)d_in[0];
  const float* WQ = (const float*)d_in[1];
  const float* bQ = (const float*)d_in[2];
  const float* WK = (const float*)d_in[3];
  const float* bK = (const float*)d_in[4];
  const float* WV = (const float*)d_in[5];
  const float* bV = (const float*)d_in[6];
  float* out = (float*)d_out;
  __bf16* wsbf = (__bf16*)d_ws;

  float* fbase = (float*)((char*)d_ws + FP_OFF);
  float* pO = fbase;                               // [NSPLIT][SEQ][DKK]
  float* pM = pO + (size_t)NSPLIT * NSD;           // [NSPLIT][SEQ]
  float* pL = pM + (size_t)NSPLIT * SEQ;           // [NSPLIT][SEQ]

  // 0) transpose + hi/lo split weights
  dim3 g0((DM * DKK) / 256, 3);
  wsplit_kernel<<<g0, 256, 0, stream>>>(WQ, WK, WV, wsbf);

  // 1) projections -> Qh/Ql, Kh/Kl (row-major), Vt (transposed)
  dim3 g1(SEQ / 16, 3);
  proj_kernel<<<g1, 128, 0, stream>>>(x, bQ, bK, bV, wsbf);

  // 2) flash attention partials: 64 q-blocks x NSPLIT KV segments
  dim3 g2(SEQ / 16 / 8, NSPLIT);
  attn_kernel<<<g2, 256, 0, stream>>>(wsbf, pO, pM, pL);

  // 3) combine partials -> fp32 output
  combine_kernel<<<(SEQ * DKK) / 256, 256, 0, stream>>>(pO, pM, pL, out);
}
